// MambaLayer_85650237817564
// MI455X (gfx1250) — matrix operand order repaired and checked by reading, not compiled, not run on hardware
//
#include <hip/hip_runtime.h>
#include <hip/hip_bf16.h>
#include <math.h>

// ---------------------------------------------------------------------------
// Mamba2 layer for MI455X (gfx1250), wave32 + WMMA bf16 + async-to-LDS DMA.
// Problem constants (from reference setup_inputs):
//   B=2, L=4096, DM=1024, d_inner=2048, n_heads=32, head_dim=64,
//   d_state=128, conv k=4, chunk=64 -> 64 chunks/batch.
// ---------------------------------------------------------------------------

#define BATCH   2
#define SEQ     4096
#define DM      1024
#define DIN     4384           // d_in_proj
#define DINPAD  4480           // padded to multiple of 128 for GEMM1 N
#define DI      2048           // d_inner
#define NH      32
#define HD      64
#define DS      128            // d_state
#define CONVD   2304           // d_inner + 2*d_state
#define CH      64             // chunk size
#define NC      64             // chunks per batch
#define BL      (BATCH*SEQ)    // 8192 rows

typedef __attribute__((ext_vector_type(16))) __bf16 bf16x16;
typedef __attribute__((ext_vector_type(8)))  __bf16 bf16x8;
typedef __attribute__((ext_vector_type(8)))  float  f32x8;

// ---------------- WMMA helpers (CDNA5 16x16x32 bf16, f32 accum) -------------

__device__ __forceinline__ f32x8 wmma_bf16(bf16x16 a, bf16x16 b, f32x8 c) {
#if defined(__HIP_DEVICE_COMPILE__)
  return __builtin_amdgcn_wmma_f32_16x16x32_bf16(
      /*neg_a=*/false, a, /*neg_b=*/false, b,
      /*c_mod=*/(short)0, c, /*reuse_a=*/false, /*reuse_b=*/false);
#else
  return c;
#endif
}

// A fragment (16 rows x 32 K) from LDS tile stored row-major [row][k],
// ld multiple of 8 elems. ISA layout: lane = row(0..15) | khalf, VGPR0..3 =
// K(khalf*8 .. +7), VGPR4..7 = K(16+khalf*8 .. +7).
__device__ __forceinline__ bf16x16 frag_a(const __bf16* t, int ld, int lane) {
  int r  = lane & 15;
  int kh = (lane >> 4) << 3;
  const __bf16* p = t + r * ld;
  bf16x8 lo = *(const bf16x8*)(p + kh);
  bf16x8 hi = *(const bf16x8*)(p + 16 + kh);
  return __builtin_shufflevector(lo, hi, 0,1,2,3,4,5,6,7,8,9,10,11,12,13,14,15);
}

// B fragment (32 K x 16 N) from LDS tile stored [n][k] (i.e. N-major rows),
// ld multiple of 8. ISA layout: lane = n(0..15) | khalf, VGPR0..3 =
// K(khalf*8 .. +7), VGPR4..7 = K(16+khalf*8 .. +7).
__device__ __forceinline__ bf16x16 frag_b(const __bf16* t, int ld, int lane) {
  int n  = lane & 15;
  int kb = (lane >> 4) << 3;
  const __bf16* p = t + n * ld + kb;
  bf16x8 lo = *(const bf16x8*)(p);
  bf16x8 hi = *(const bf16x8*)(p + 16);
  return __builtin_shufflevector(lo, hi, 0,1,2,3,4,5,6,7,8,9,10,11,12,13,14,15);
}

// ---------------- CDNA5 async global->LDS DMA (ASYNCcnt tracked) ------------
// IOFFSET adds to BOTH lds and global address, so one VGPR addr pair serves
// both 16-byte halves of a 32-byte row segment.

__device__ __forceinline__ unsigned lds_addr32(const void* p) {
  // LDS generic addresses truncate to the wave-relative LDS offset (ISA 10.2).
  return (unsigned)(size_t)p;
}

__device__ __forceinline__ void async_copy_b128x2(unsigned lds, const void* g) {
#if defined(__HIP_DEVICE_COMPILE__)
  asm volatile("global_load_async_to_lds_b128 %0, %1, off\n\t"
               "global_load_async_to_lds_b128 %0, %1, off offset:16"
               :
               : "v"(lds), "v"(g)
               : "memory");
#endif
}

__device__ __forceinline__ void wait_async0() {
#if defined(__HIP_DEVICE_COMPILE__)
  asm volatile("s_wait_asynccnt 0x0" ::: "memory");
#endif
}

__device__ __forceinline__ float siluf(float v) {
  return v / (1.0f + __expf(-v));
}

// ---------------- conversion kernels ---------------------------------------

__global__ void k_f32_to_bf16(const float* __restrict__ in,
                              __bf16* __restrict__ out, size_t n) {
  size_t i = (size_t)blockIdx.x * blockDim.x + threadIdx.x;
  if (i < n) out[i] = (__bf16)in[i];
}

// in: [K][N] f32 row-major; out: [Npad][K] bf16 (transposed, zero padded rows)
__global__ void k_transpose_bf16(const float* __restrict__ in,
                                 __bf16* __restrict__ out,
                                 int K, int N, int Npad) {
  size_t i = (size_t)blockIdx.x * blockDim.x + threadIdx.x;
  size_t total = (size_t)Npad * K;
  if (i >= total) return;
  int n = (int)(i / K);
  int k = (int)(i % K);
  float v = (n < N) ? in[(size_t)k * N + n] : 0.0f;
  out[i] = (__bf16)v;
}

// ---------------- bf16 WMMA GEMM: C[M x ldc](f32) = A[MxK] * B^T[NxK] -------
// block = 256 threads (8 waves). Tile 128(M) x 128(N), BK = 32, software
// pipelined (2 K-steps/iteration, compile-time buffer indices) via
// GLOBAL_LOAD_ASYNC_TO_LDS_B128: no staging VGPRs, DMA for tile t+1 overlaps
// the 8 WMMAs of tile t; one s_wait_asynccnt + one barrier per K-step.
// wave (wm 0..3, wn 0..1) owns 32x64 -> 2x4 wmma accumulators.
// Requires K % 64 == 0 (true for K = 1024, 2048).

__global__ void k_gemm_bf16(const __bf16* __restrict__ A,
                            const __bf16* __restrict__ BT,
                            float* __restrict__ C,
                            int K, int ldc) {
  __shared__ __bf16 sA[2][128][40];
  __shared__ __bf16 sB[2][128][40];
  const int tid  = threadIdx.x;
  const int lane = tid & 31;
  const int w    = tid >> 5;
  const int wm   = w & 3;
  const int wn   = w >> 2;
  const size_t bm = (size_t)blockIdx.y * 128;
  const size_t bn = (size_t)blockIdx.x * 128;

  f32x8 acc[2][4];
#pragma unroll
  for (int i = 0; i < 2; ++i)
#pragma unroll
    for (int j = 0; j < 4; ++j) acc[i][j] = (f32x8){0,0,0,0,0,0,0,0};

  // per-thread DMA assignment: 32 bytes of A and 32 bytes of B per K-step
  const int lrow = tid >> 1;
  const int lcol = (tid & 1) * 16;
  const __bf16* ga = A  + (bm + lrow) * K + lcol;
  const __bf16* gb = BT + (bn + lrow) * K + lcol;
  const unsigned la0 = lds_addr32(&sA[0][lrow][lcol]);
  const unsigned la1 = lds_addr32(&sA[1][lrow][lcol]);
  const unsigned lb0 = lds_addr32(&sB[0][lrow][lcol]);
  const unsigned lb1 = lds_addr32(&sB[1][lrow][lcol]);

  // one K-step of WMMA work from buffer `bufc` (bufc is a constant 0/1)
  auto compute = [&](int bufc) {
    bf16x16 af[2], bfg[4];
#pragma unroll
    for (int i = 0; i < 2; ++i)
      af[i] = frag_a(&sA[bufc][wm * 32 + i * 16][0], 40, lane);
#pragma unroll
    for (int j = 0; j < 4; ++j)
      bfg[j] = frag_b(&sB[bufc][wn * 64 + j * 16][0], 40, lane);
#pragma unroll
    for (int i = 0; i < 2; ++i)
#pragma unroll
      for (int j = 0; j < 4; ++j)
        acc[i][j] = wmma_bf16(af[i], bfg[j], acc[i][j]);
  };

  // prologue: DMA K-tile 0 into buffer 0
  async_copy_b128x2(la0, ga);
  async_copy_b128x2(lb0, gb);
  ga += 32;
  gb += 32;
  wait_async0();
  __syncthreads();

  const int nsteps = K >> 6;   // pairs of 32-wide K-tiles
  for (int s = 0; s < nsteps; ++s) {
    // DMA tile (2s+1) into buffer 1, overlapping compute on buffer 0
    async_copy_b128x2(la1, ga);
    async_copy_b128x2(lb1, gb);
    ga += 32;
    gb += 32;
    compute(0);
    wait_async0();
    __syncthreads();

    // DMA tile (2s+2) into buffer 0, overlapping compute on buffer 1
    if (s + 1 < nsteps) {
      async_copy_b128x2(la0, ga);
      async_copy_b128x2(lb0, gb);
      ga += 32;
      gb += 32;
    }
    compute(1);
    wait_async0();
    __syncthreads();
  }

  const int moff = (lane >> 4) * 8;
  const int ncol = lane & 15;
#pragma unroll
  for (int i = 0; i < 2; ++i)
#pragma unroll
    for (int j = 0; j < 4; ++j) {
      size_t mbase = bm + wm * 32 + i * 16 + moff;
      size_t nn    = bn + wn * 64 + j * 16 + ncol;
#pragma unroll
      for (int r = 0; r < 8; ++r)
        C[(mbase + r) * (size_t)ldc + nn] = acc[i][j][r];
    }
}

// ---------------- elementwise stages ----------------------------------------

// dt = softplus(zx[:, 4352+h] + dt_bias[h])
__global__ void k_dt_softplus(const float* __restrict__ zx,
                              const float* __restrict__ dt_bias,
                              float* __restrict__ dt) {
  int i = blockIdx.x * blockDim.x + threadIdx.x;
  if (i >= BL * NH) return;
  int row = i >> 5, h = i & 31;
  float x = zx[(size_t)row * DINPAD + (DI + CONVD) + h] + dt_bias[h];
  dt[i] = (x > 20.0f) ? x : log1pf(__expf(x));
}

// causal depthwise conv (k=4) + bias + silu over zx cols [DI, DI+CONVD)
__global__ void k_conv_silu(const float* __restrict__ zx,
                            const float* __restrict__ w,
                            const float* __restrict__ bias,
                            float* __restrict__ xbc) {
  size_t i = (size_t)blockIdx.x * blockDim.x + threadIdx.x;
  if (i >= (size_t)BL * CONVD) return;
  int row = (int)(i / CONVD);
  int ch  = (int)(i % CONVD);
  int l   = row & (SEQ - 1);
  float a = bias[ch];
#pragma unroll
  for (int j = 0; j < 4; ++j) {
    int ls = l - 3 + j;
    if (ls >= 0)
      a += w[j * CONVD + ch] * zx[(size_t)(row - 3 + j) * DINPAD + DI + ch];
  }
  xbc[i] = siluf(a);
}

// per (b,c,h): acs[l] = cumsum_l( -exp(A_log[h]) * dt[b, c*64+l, h] )
__global__ void k_chunk_cumsum(const float* __restrict__ dt,
                               const float* __restrict__ A_log,
                               float* __restrict__ acs) {
  int t = blockIdx.x * blockDim.x + threadIdx.x;
  if (t >= BATCH * NC * NH) return;
  int b = t / (NC * NH);
  int c = (t / NH) % NC;
  int h = t % NH;
  float nega = -__expf(A_log[h]);
  float s = 0.0f;
  size_t rbase = ((size_t)b * SEQ + c * CH);
  size_t obase = (size_t)t * CH;   // t == ((b*NC + c)*NH + h)
  for (int l = 0; l < CH; ++l) {
    s += nega * dt[(rbase + l) * NH + h];
    acs[obase + l] = s;
  }
}

// ---------------- SSD per-chunk WMMA kernel ---------------------------------
// grid (NH, NC, BATCH), 256 threads. Computes Y_diag (+ D*xs) and chunk
// states via WMMA bf16. Shared regions reused: sC region becomes BTd region.

__global__ void k_ssd_chunk(const float* __restrict__ xbc,
                            const float* __restrict__ dt,
                            const float* __restrict__ acs,
                            const float* __restrict__ Dvec,
                            float* __restrict__ Y,
                            float* __restrict__ states) {
  const int h = blockIdx.x, c = blockIdx.y, b = blockIdx.z;
  const int tid = threadIdx.x, lane = tid & 31, w = tid >> 5;

  __shared__ alignas(16) char smem[49408];
  __bf16* sC   = (__bf16*)(smem);               // 64 x 128 (C chunk)
  __bf16* sB   = (__bf16*)(smem + 16384);       // 64 x 128 (B chunk)
  __bf16* sXT  = (__bf16*)(smem + 32768);       // 64 x 64  (X^T: [p][l])
  __bf16* sM   = (__bf16*)(smem + 40960);       // 64 x 64  (masked CB: [l][s])
  float*  sAcs = (float*)(smem + 49152);        // 64
  __bf16* sBTd = sC;                            // reuse: 128 x 64 ([n][l])

  const size_t rbase = (size_t)b * SEQ + c * CH;
  const size_t abase = ((size_t)((b * NC + c) * NH) + h) * CH;

  // ---- phase A: stage C, B, X^T, acs -------------------------------------
  for (int i = tid; i < CH * DS; i += 256) {
    int l = i >> 7, n = i & 127;
    size_t g = (rbase + l) * CONVD;
    sB[l * 128 + n] = (__bf16)xbc[g + DI + n];
    sC[l * 128 + n] = (__bf16)xbc[g + DI + DS + n];
  }
  for (int i = tid; i < CH * HD; i += 256) {
    int p = i >> 6, l = i & 63;
    float v = xbc[(rbase + l) * CONVD + h * HD + p] * dt[(rbase + l) * NH + h];
    sXT[p * 64 + l] = (__bf16)v;
  }
  if (tid < CH) sAcs[tid] = acs[abase + tid];
  __syncthreads();

  const int moff = (lane >> 4) * 8;
  const int ncol = lane & 15;

  // ---- phase 1: M = (C * B^T) o L  (16 tiles 16x16, K = 128) -------------
  for (int t = w; t < 16; t += 8) {
    int tm = t >> 2, tn = t & 3;
    f32x8 acc = (f32x8){0,0,0,0,0,0,0,0};
#pragma unroll
    for (int kk = 0; kk < 4; ++kk) {
      bf16x16 af = frag_a(sC + tm * 16 * 128 + kk * 32, 128, lane);
      bf16x16 bf = frag_b(sB + tn * 16 * 128 + kk * 32, 128, lane);
      acc = wmma_bf16(af, bf, acc);
    }
#pragma unroll
    for (int r = 0; r < 8; ++r) {
      int m = tm * 16 + moff + r;
      int s = tn * 16 + ncol;
      float v = (s <= m) ? acc[r] * __expf(sAcs[m] - sAcs[s]) : 0.0f;
      sM[m * 64 + s] = (__bf16)v;
    }
  }
  __syncthreads();

  // ---- phase 2: Y_diag = M @ X  (+ D[h]*xs) ------------------------------
  const float Dh = Dvec[h];
  for (int t = w; t < 16; t += 8) {
    int tl = t >> 2, tp = t & 3;
    f32x8 acc = (f32x8){0,0,0,0,0,0,0,0};
#pragma unroll
    for (int kk = 0; kk < 2; ++kk) {
      bf16x16 af = frag_a(sM  + tl * 16 * 64 + kk * 32, 64, lane);
      bf16x16 bf = frag_b(sXT + tp * 16 * 64 + kk * 32, 64, lane);
      acc = wmma_bf16(af, bf, acc);
    }
#pragma unroll
    for (int r = 0; r < 8; ++r) {
      int l = tl * 16 + moff + r;
      int p = tp * 16 + ncol;
      size_t g = (rbase + l);
      float xs = xbc[g * CONVD + h * HD + p];
      Y[g * DI + h * HD + p] = acc[r] + Dh * xs;
    }
  }

  // ---- phase 3: BTd[n][l] = B[l][n] * exp(acs[63] - acs[l]) --------------
  for (int i = tid; i < DS * CH; i += 256) {
    int n = i >> 6, l = i & 63;
    float dec = __expf(sAcs[CH - 1] - sAcs[l]);
    sBTd[n * 64 + l] = (__bf16)((float)sB[l * 128 + n] * dec);
  }
  __syncthreads();

  // ---- phase 4: state[p][n] = sum_l XT[p][l] * BTd[n][l] (32 tiles) ------
  size_t sbase = ((size_t)((b * NC + c) * NH) + h) * (HD * DS);
  for (int t = w * 4; t < w * 4 + 4; ++t) {
    int tp = t >> 3, tn = t & 7;
    f32x8 acc = (f32x8){0,0,0,0,0,0,0,0};
#pragma unroll
    for (int kk = 0; kk < 2; ++kk) {
      bf16x16 af = frag_a(sXT  + tp * 16 * 64 + kk * 32, 64, lane);
      bf16x16 bf = frag_b(sBTd + tn * 16 * 64 + kk * 32, 64, lane);
      acc = wmma_bf16(af, bf, acc);
    }
#pragma unroll
    for (int r = 0; r < 8; ++r) {
      int p = tp * 16 + moff + r;
      int n = tn * 16 + ncol;
      states[sbase + (size_t)p * DS + n] = acc[r];
    }
  }
}

// ---------------- inter-chunk sequential scan (in place) --------------------
// states[b,c,h,p,n] : overwrite with prefix state S_c; S <- e^{sumA_c}*S + st_c

__global__ void k_chunk_scan(float* __restrict__ states,
                             const float* __restrict__ acs) {
  int b = blockIdx.x >> 5, h = blockIdx.x & 31;
  float S[32];
#pragma unroll
  for (int i = 0; i < 32; ++i) S[i] = 0.0f;
  for (int c = 0; c < NC; ++c) {
    float lam = __expf(acs[(((size_t)(b * NC + c) * NH) + h) * CH + (CH - 1)]);
    size_t base = ((size_t)((b * NC + c) * NH) + h) * (HD * DS);
#pragma unroll
    for (int i = 0; i < 32; ++i) {
      size_t e = base + threadIdx.x + i * 256;
      float tmp = states[e];
      states[e] = S[i];
      S[i] = lam * S[i] + tmp;
    }
  }
}

// ---------------- Y_off = C @ S_prev^T scaled by exp(acs[l]) ----------------

__global__ void k_ssd_yoff(const float* __restrict__ xbc,
                           const float* __restrict__ acs,
                           const float* __restrict__ states,
                           float* __restrict__ Y) {
  const int h = blockIdx.x, c = blockIdx.y, b = blockIdx.z;
  const int tid = threadIdx.x, lane = tid & 31, w = tid >> 5;

  __shared__ alignas(16) char smem[33024];
  __bf16* sC   = (__bf16*)(smem);             // 64 x 128 (C chunk [l][n])
  __bf16* sS   = (__bf16*)(smem + 16384);     // 64 x 128 (S_prev [p][n])
  float*  sAcs = (float*)(smem + 32768);

  const size_t rbase = (size_t)b * SEQ + c * CH;
  const size_t abase = ((size_t)((b * NC + c) * NH) + h) * CH;
  const size_t sbase = ((size_t)((b * NC + c) * NH) + h) * (HD * DS);

  for (int i = tid; i < CH * DS; i += 256) {
    int l = i >> 7, n = i & 127;
    sC[l * 128 + n] = (__bf16)xbc[(rbase + l) * CONVD + DI + DS + n];
    sS[l * 128 + n] = (__bf16)states[sbase + i];   // here l plays role of p
  }
  if (tid < CH) sAcs[tid] = acs[abase + tid];
  __syncthreads();

  const int moff = (lane >> 4) * 8;
  const int ncol = lane & 15;
  for (int t = w; t < 16; t += 8) {
    int tl = t >> 2, tp = t & 3;
    f32x8 acc = (f32x8){0,0,0,0,0,0,0,0};
#pragma unroll
    for (int kk = 0; kk < 4; ++kk) {
      bf16x16 af = frag_a(sC + tl * 16 * 128 + kk * 32, 128, lane);
      bf16x16 bf = frag_b(sS + tp * 16 * 128 + kk * 32, 128, lane);
      acc = wmma_bf16(af, bf, acc);
    }
#pragma unroll
    for (int r = 0; r < 8; ++r) {
      int l = tl * 16 + moff + r;
      int p = tp * 16 + ncol;
      size_t g = (rbase + l) * DI + h * HD + p;
      Y[g] += __expf(sAcs[l]) * acc[r];
    }
  }
}

// ---------------- gating + RMS norm -> bf16 ---------------------------------

__global__ void k_gate_rmsnorm(const float* __restrict__ Y,
                               const float* __restrict__ zx,
                               const float* __restrict__ scale,
                               __bf16* __restrict__ out) {
  const int row = blockIdx.x;
  const int tid = threadIdx.x;
  __shared__ float red[256];
  float yz[8];
  float ss = 0.0f;
#pragma unroll
  for (int j = 0; j < 8; ++j) {
    int col = tid + j * 256;
    float y = Y[(size_t)row * DI + col];
    float z = zx[(size_t)row * DINPAD + col];
    float v = y * siluf(z);
    yz[j] = v;
    ss += v * v;
  }
  red[tid] = ss;
  __syncthreads();
  for (int s = 128; s > 0; s >>= 1) {
    if (tid < s) red[tid] += red[tid + s];
    __syncthreads();
  }
  float r = rsqrtf(red[0] / (float)DI + 1e-6f);
#pragma unroll
  for (int j = 0; j < 8; ++j) {
    int col = tid + j * 256;
    out[(size_t)row * DI + col] = (__bf16)(yz[j] * r * scale[col]);
  }
}

// ---------------------------------------------------------------------------
// launch
// ---------------------------------------------------------------------------

extern "C" void kernel_launch(void* const* d_in, const int* in_sizes, int n_in,
                              void* d_out, int out_size, void* d_ws, size_t ws_size,
                              hipStream_t stream) {
  const float* x        = (const float*)d_in[0];
  const float* w1       = (const float*)d_in[1];   // [1024][4384]
  const float* conv_w   = (const float*)d_in[2];   // [4][1][2304]
  const float* conv_b   = (const float*)d_in[3];
  const float* dt_bias  = (const float*)d_in[4];
  const float* A_log    = (const float*)d_in[5];
  const float* Dvec     = (const float*)d_in[6];
  const float* nscale   = (const float*)d_in[7];
  const float* w2       = (const float*)d_in[8];   // [2048][1024]
  float* out            = (float*)d_out;

  // workspace carve-up (~489 MB)
  char* ws = (char*)d_ws;
  size_t off = 0;
  auto alloc = [&](size_t bytes) {
    size_t o = off;
    off += (bytes + 255) & ~(size_t)255;
    return o;
  };
  __bf16* xb16 = (__bf16*)(ws + alloc((size_t)BL * DM * 2));
  __bf16* w1T  = (__bf16*)(ws + alloc((size_t)DINPAD * DM * 2));
  __bf16* w2T  = (__bf16*)(ws + alloc((size_t)DM * DI * 2));
  float*  zx   = (float*)(ws + alloc((size_t)BL * DINPAD * 4));
  float*  dt   = (float*)(ws + alloc((size_t)BL * NH * 4));
  float*  xbc  = (float*)(ws + alloc((size_t)BL * CONVD * 4));
  float*  acs  = (float*)(ws + alloc((size_t)BATCH * NC * NH * CH * 4));
  float*  st   = (float*)(ws + alloc((size_t)BATCH * NC * NH * HD * DS * 4));
  float*  Ybuf = (float*)(ws + alloc((size_t)BL * DI * 4));
  __bf16* yb16 = (__bf16*)(ws + alloc((size_t)BL * DI * 2));

  // 1. precision conversion
  {
    size_t n = (size_t)BL * DM;
    k_f32_to_bf16<<<(unsigned)((n + 255) / 256), 256, 0, stream>>>(x, xb16, n);
  }
  {
    size_t n = (size_t)DINPAD * DM;
    k_transpose_bf16<<<(unsigned)((n + 255) / 256), 256, 0, stream>>>(
        w1, w1T, DM, DIN, DINPAD);
  }
  {
    size_t n = (size_t)DM * DI;
    k_transpose_bf16<<<(unsigned)((n + 255) / 256), 256, 0, stream>>>(
        w2, w2T, DI, DM, DM);
  }

  // 2. in_proj GEMM: zx[8192 x 4480] = x @ W1  (WMMA bf16, async DMA)
  k_gemm_bf16<<<dim3(DINPAD / 128, BL / 128), 256, 0, stream>>>(
      xb16, w1T, zx, DM, DINPAD);

  // 3. dt = softplus, depthwise conv + silu
  k_dt_softplus<<<(BL * NH + 255) / 256, 256, 0, stream>>>(zx, dt_bias, dt);
  {
    size_t n = (size_t)BL * CONVD;
    k_conv_silu<<<(unsigned)((n + 255) / 256), 256, 0, stream>>>(
        zx, conv_w, conv_b, xbc);
  }

  // 4. per-chunk cumulative decay
  k_chunk_cumsum<<<(BATCH * NC * NH + 255) / 256, 256, 0, stream>>>(
      dt, A_log, acs);

  // 5. SSD: diag + chunk states (WMMA), sequential inter-chunk scan, off-diag
  k_ssd_chunk<<<dim3(NH, NC, BATCH), 256, 0, stream>>>(
      xbc, dt, acs, Dvec, Ybuf, st);
  k_chunk_scan<<<BATCH * NH, 256, 0, stream>>>(st, acs);
  k_ssd_yoff<<<dim3(NH, NC, BATCH), 256, 0, stream>>>(xbc, acs, st, Ybuf);

  // 6. gate with silu(z), RMS norm, convert to bf16
  k_gate_rmsnorm<<<BL, 256, 0, stream>>>(Ybuf, zx, nscale, yb16);

  // 7. out_proj GEMM: out[8192 x 1024] = y @ W2 (WMMA bf16, async DMA)
  k_gemm_bf16<<<dim3(DM / 128, BL / 128), 256, 0, stream>>>(
      yb16, w2T, out, DI, DM);

  (void)in_sizes; (void)n_in; (void)out_size; (void)ws_size;
}
